// MFFNet8_67834713473663
// MI455X (gfx1250) — matrix operand order repaired and checked by reading, not compiled, not run on hardware
//
#include <hip/hip_runtime.h>
#include <hip/hip_bf16.h>
#include <string.h>

typedef __attribute__((ext_vector_type(16))) __bf16   v16bf;
typedef __attribute__((ext_vector_type(8)))  float    v8f;
typedef __attribute__((ext_vector_type(8)))  unsigned v8u;

constexpr int B_1X1 = 0, B_TCONV = 1, B_VLAST = 2, B_EINSUM = 3;
constexpr int OUT_ROW = 0, OUT_COL = 1;

struct GemmP {
  const float* W;
  const float* bias;
  const float* bnscale;
  const float* bnshift;
  const float* X;
  float*       Out;
  int M, K, Cols;
  int xStrideN, oStrideN, wStrideN;   // element counts < 2^31 -> 32-bit math
  int wTrans;
  int bMode;
  int ldB;               // B_VLAST: addr = col*ldB + k
  int Tin, strideT;      // B_TCONV (k=3, pad=1, dil=1, V=18)
  int Tlen, cmid;        // B_EINSUM
  int outMode, oLd;
  int relu, accum;
};

// pack two f32 -> packed bf16 pair (RNE): 2x v_add3 + 1x v_perm_b32
__device__ __forceinline__ unsigned pk2bf(float f0, float f1) {
  unsigned u0 = __builtin_bit_cast(unsigned, f0);
  unsigned u1 = __builtin_bit_cast(unsigned, f1);
  u0 = u0 + 0x7FFFu + ((u0 >> 16) & 1u);
  u1 = u1 + 0x7FFFu + ((u1 >> 16) & 1u);
  return __builtin_amdgcn_perm(u1, u0, 0x07060302u);  // {u1.hi16, u0.hi16}
}

struct ColD { int colC, bt, bv, bc; };

// ---- A element: clamp only (garbage rows feed discarded outputs; K-tail is
// zeroed on the B side, so A needs no zero select at all).
template <int WT, bool TAIL>
__device__ __forceinline__ float load_a_elem(const GemmP& p, const float* W,
                                             int kk, int arow) {
  int kkc = TAIL ? ((kk < p.K) ? kk : p.K - 1) : kk;
  int addr = WT ? (kkc * p.M + arow) : (arow * p.K + kkc);
  return W[addr];
}

template <int WT, bool TAIL>
__device__ __forceinline__ v16bf load_a_frag(const GemmP& p, const float* W,
                                             int k0, int lhalf, int arow) {
  v8u pk;
#pragma unroll
  for (int jj = 0; jj < 8; ++jj) {
    int j0 = 2 * jj;                                   // pairs never straddle j=7/8
    int kk0 = k0 + lhalf * 8 + ((j0 < 8) ? j0 : j0 + 8);
    float w0 = load_a_elem<WT, TAIL>(p, W, kk0, arow);
    float w1 = load_a_elem<WT, TAIL>(p, W, kk0 + 1, arow);
    pk[jj] = pk2bf(w0, w1);
  }
  return __builtin_bit_cast(v16bf, pk);
}

// ---- B element: address always clamped in-bounds; zero only for K-tail and
// TCONV time padding (those affect *valid* outputs).
template <int BM, bool TAIL>
__device__ __forceinline__ float load_b_elem(const GemmP& p, const float* X,
                                             int kk, const ColD& cd) {
  bool zero = false;
  int kkc = kk;
  if (TAIL) { zero = kk >= p.K; kkc = zero ? (p.K - 1) : kk; }
  int addr = 0;
  if (BM == B_1X1)   addr = kkc * p.Cols + cd.colC;
  if (BM == B_VLAST) addr = cd.colC * p.ldB + kkc;
  if (BM == B_TCONV) {
    int ci = kkc / 3, dt = kkc - ci * 3;
    int tin = cd.bt * p.strideT + dt - 1;
    zero = zero | (tin < 0) | (tin >= p.Tin);
    int tc = tin < 0 ? 0 : (tin >= p.Tin ? p.Tin - 1 : tin);
    addr = (ci * p.Tin + tc) * 18 + cd.bv;
  }
  if (BM == B_EINSUM) {
    int ak = kkc / 18, vv = kkc - ak * 18;
    addr = ((ak * p.cmid + cd.bc) * p.Tlen + cd.bt) * 18 + vv;
  }
  float xv = X[addr];
  return zero ? 0.f : xv;
}

template <int BM, bool TAIL>
__device__ __forceinline__ v16bf load_b_frag(const GemmP& p, const float* X,
                                             int k0, int lhalf, const ColD& cd) {
  v8u pk;
#pragma unroll
  for (int jj = 0; jj < 8; ++jj) {
    int kk0 = k0 + lhalf * 8 + ((2 * jj < 8) ? 2 * jj : 2 * jj + 8);
    float x0 = load_b_elem<BM, TAIL>(p, X, kk0, cd);
    float x1 = load_b_elem<BM, TAIL>(p, X, kk0 + 1, cd);
    pk[jj] = pk2bf(x0, x1);
  }
  return __builtin_bit_cast(v16bf, pk);
}

// ---------------------------------------------------------------------------
// WMMA GEMM. 128 threads = 4 waves; each wave computes a 16x32 output tile
// (two 16x16 WMMAs sharing one A fragment). Main K loop has no selects at
// all; one uniform branch handles the K tail. EXEC all-ones at every WMMA.
// ---------------------------------------------------------------------------
template <int BM, int WT>
__global__ void k_gemm(GemmP p) {
  const int lane  = threadIdx.x & 31;
  const int wave  = threadIdx.x >> 5;
  const int lhalf = lane >> 4;
  const int l16   = lane & 15;
  const int m0    = blockIdx.y * 16;
  const int nz    = blockIdx.z;
  const int colA  = blockIdx.x * 128 + wave * 32 + l16;
  const int colB  = colA + 16;

  const float* X  = p.X + (size_t)nz * (size_t)p.xStrideN;
  const float* W  = p.W + (size_t)nz * (size_t)p.wStrideN;
  float*      Out = p.Out + (size_t)nz * (size_t)p.oStrideN;

  auto mkcol = [&](int col) {
    ColD d; d.bt = 0; d.bv = 0; d.bc = 0;
    d.colC = (col < p.Cols) ? col : (p.Cols - 1);
    if (BM == B_TCONV)  { d.bt = d.colC / 18;     d.bv = d.colC - d.bt * 18; }
    if (BM == B_EINSUM) { d.bc = d.colC / p.Tlen; d.bt = d.colC - d.bc * p.Tlen; }
    return d;
  };
  const ColD cd0 = mkcol(colA), cd1 = mkcol(colB);
  const bool c0ok = colA < p.Cols, c1ok = colB < p.Cols;

  const int  am   = m0 + l16;
  const bool amok = am < p.M;
  const int  arow = amok ? am : (p.M - 1);

  v8f acc0 = {0.f,0.f,0.f,0.f,0.f,0.f,0.f,0.f};
  v8f acc1 = {0.f,0.f,0.f,0.f,0.f,0.f,0.f,0.f};

  const int kMain = p.K & ~31;
  for (int k0 = 0; k0 < kMain; k0 += 32) {
    v16bf a  = load_a_frag<WT, false>(p, W, k0, lhalf, arow);
    v16bf b0 = load_b_frag<BM, false>(p, X, k0, lhalf, cd0);
    v16bf b1 = load_b_frag<BM, false>(p, X, k0, lhalf, cd1);
    if (BM == B_1X1 && (k0 + 64) <= p.K)
      __builtin_prefetch(&X[(k0 + 32 + lhalf * 16) * p.Cols + cd0.colC], 0, 1);
    acc0 = __builtin_amdgcn_wmma_f32_16x16x32_bf16(false, a, false, b0, (short)0, acc0,
                                                   false, false);
    acc1 = __builtin_amdgcn_wmma_f32_16x16x32_bf16(false, a, false, b1, (short)0, acc1,
                                                   false, false);
  }
  if (kMain < p.K) {  // uniform scalar branch
    v16bf a  = load_a_frag<WT, true>(p, W, kMain, lhalf, arow);
    v16bf b0 = load_b_frag<BM, true>(p, X, kMain, lhalf, cd0);
    v16bf b1 = load_b_frag<BM, true>(p, X, kMain, lhalf, cd1);
    acc0 = __builtin_amdgcn_wmma_f32_16x16x32_bf16(false, a, false, b0, (short)0, acc0,
                                                   false, false);
    acc1 = __builtin_amdgcn_wmma_f32_16x16x32_bf16(false, a, false, b1, (short)0, acc1,
                                                   false, false);
  }

  // epilogue: C layout VGPR r = row r (lanes 0-15) / r+8 (lanes 16-31)
#pragma unroll
  for (int r = 0; r < 8; ++r) {
    int m = m0 + r + lhalf * 8;
    bool mok = m < p.M;
    int mc = mok ? m : 0;
    float badd = p.bias    ? p.bias[mc]    : 0.f;
    float bsc  = p.bnscale ? p.bnscale[mc] : 1.f;
    float bsh  = p.bnscale ? p.bnshift[mc] : 0.f;
    float v0 = (acc0[r] + badd) * bsc + bsh;
    float v1 = (acc1[r] + badd) * bsc + bsh;
    if (p.relu) { v0 = fmaxf(v0, 0.f); v1 = fmaxf(v1, 0.f); }
    int oa0 = (p.outMode == OUT_ROW) ? (m * p.Cols + colA) : (colA * p.oLd + m);
    int oa1 = (p.outMode == OUT_ROW) ? (m * p.Cols + colB) : (colB * p.oLd + m);
    if (mok && c0ok) { if (p.accum) Out[oa0] += v0; else Out[oa0] = v0; }
    if (mok && c1ok) { if (p.accum) Out[oa1] += v1; else Out[oa1] = v1; }
  }
}

// ---------------------------------------------------------------------------
// Elementwise kernels: channel -> blockIdx.y, batch -> blockIdx.z
// ---------------------------------------------------------------------------
__global__ void k_bn_prepare(const float* g, const float* b, const float* m,
                             const float* v, float* scale, float* shift, int C) {
  int i = blockIdx.x * blockDim.x + threadIdx.x;
  if (i < C) {
    float s = g[i] * rsqrtf(v[i] + 1e-5f);
    scale[i] = s;
    shift[i] = b[i] - m[i] * s;
  }
}

__global__ void k_input_bn(const float* x, const float* sc, const float* sh, float* out) {
  int i = blockIdx.x * 256 + threadIdx.x;
  if (i >= 5400) return;
  int c = blockIdx.y, nb = blockIdx.z;
  int v = i % 18, t = i / 18;
  int n0 = nb >> 1, m = nb & 1;
  int ch = v * 3 + c;
  size_t in_idx = ((((size_t)n0 * 3 + c) * 300 + t) * 18 + v) * 2 + m;
  out[((size_t)nb * 3 + c) * 5400 + i] = x[in_idx] * sc[ch] + sh[ch];
}

__global__ void k_bnapply(const float* in, float* out, const float* sc, const float* sh,
                          int TV) {
  int t = blockIdx.x * 256 + threadIdx.x;
  if (t >= TV) return;
  int c = blockIdx.y, C = gridDim.y, n = blockIdx.z;
  size_t idx = ((size_t)n * C + c) * TV + t;
  out[idx] = in[idx] * sc[c] + sh[c];
}

__global__ void k_bn_add_relu(const float* e, const float* res, float* out,
                              const float* sc, const float* sh, int TV) {
  int t = blockIdx.x * 256 + threadIdx.x;
  if (t >= TV) return;
  int c = blockIdx.y, C = gridDim.y, n = blockIdx.z;
  size_t idx = ((size_t)n * C + c) * TV + t;
  out[idx] = fmaxf(e[idx] * sc[c] + sh[c] + res[idx], 0.f);
}

__global__ void k_addshift(float* h, const float* s) {
  int i = blockIdx.x * 256 + threadIdx.x;
  if (i >= 5400) return;
  int c = blockIdx.y, nb = blockIdx.z;
  size_t idx = ((size_t)nb * 3 + c) * 5400 + i;
  if (c != 2) h[idx] += s[idx];
}

__global__ void k_motion(const float* h, float* out) {
  int i = blockIdx.x * 256 + threadIdx.x;
  if (i >= 5400) return;
  int c = blockIdx.y, nb = blockIdx.z;
  int t = i / 18;
  size_t idx = ((size_t)nb * 3 + c) * 5400 + i;
  out[idx] = (t == 0) ? 0.f : (h[idx] - h[idx - 18]);
}

__global__ void k_scaleT(const float* in, float* out, int T2, int r,
                         int istrideN, int ostrideN) {
  int i = blockIdx.x * 256 + threadIdx.x;
  if (i >= T2 * 18) return;
  int c = blockIdx.y, n = blockIdx.z;
  int v = i % 18, t2 = i / 18;
  const float* ip = in + (size_t)n * istrideN + (c * T2 * r + t2 * r) * 18 + v;
  float mx = ip[0];
  for (int q = 1; q < r; ++q) mx = fmaxf(mx, ip[q * 18]);
  out[(size_t)n * ostrideN + (c * T2 + t2) * 18 + v] = mx;
}

__global__ void k_attn_logits(const float* m, float* logits, int Kred, int mStrideN) {
  int n = blockIdx.x, tid = threadIdx.x;
  if (tid >= 324) return;
  int i = tid / 18, j = tid % 18;
  const float* mp = m + (size_t)n * mStrideN;
  float acc = 0.f;
  for (int k = 0; k < Kred; ++k) acc += mp[k * 18 + i] * mp[k * 18 + j];
  logits[n * 324 + tid] = acc / (float)Kred;
}

__global__ void k_softmax18(const float* L, float* S) {
  int n = blockIdx.x, j = threadIdx.x;
  if (j >= 18) return;
  const float* lp = L + n * 324;
  float mx = -1e30f;
  for (int i = 0; i < 18; ++i) mx = fmaxf(mx, lp[i * 18 + j]);
  float sum = 0.f;
  for (int i = 0; i < 18; ++i) sum += expf(lp[i * 18 + j] - mx);
  float inv = 1.f / sum;
  for (int i = 0; i < 18; ++i) S[n * 324 + i * 18 + j] = expf(lp[i * 18 + j] - mx) * inv;
}

__global__ void k_maxT(const float* in, float* out, int T) {
  int v = threadIdx.x;
  if (v >= 18) return;
  int c = blockIdx.y, C = gridDim.y, n = blockIdx.z;
  const float* ip = in + (((size_t)n * C + c) * T) * 18 + v;
  float mx = ip[0];
  for (int t = 1; t < T; ++t) mx = fmaxf(mx, ip[t * 18]);
  out[((size_t)n * C + c) * 18 + v] = mx;
}

__global__ void k_pool_mv(const float* y2, float* out) {
  int i = blockIdx.x * blockDim.x + threadIdx.x;
  if (i >= 32 * 256) return;
  int c = i % 256, n0 = i / 256;
  float mx = -1e30f;
  for (int m = 0; m < 2; ++m)
    for (int v = 0; v < 18; ++v)
      mx = fmaxf(mx, y2[(((size_t)(n0 * 2 + m)) * 256 + c) * 18 + v]);
  out[i] = mx;
}

// ---------------------------------------------------------------------------
struct BnRaw { const float *b, *g, *m, *v; };
struct BnP   { const float *scale, *shift; };

extern "C" void kernel_launch(void* const* d_in, const int* in_sizes, int n_in,
                              void* d_out, int out_size, void* d_ws, size_t ws_size,
                              hipStream_t stream) {
  (void)in_sizes; (void)n_in; (void)out_size; (void)ws_size;

  // ----- input unpack (x first, then params pytree in sorted-dict order) -----
  int ii = 0;
  auto F = [&](int) { return (const float*)d_in[ii++]; };
  const float* IN_x = F(0);
  const float* A_s1 = F(0);
  const float* A_s2 = F(0);
  const float* As[4]; for (int k = 0; k < 4; ++k) As[k] = F(0);
  const float *fus_b = F(0), *fus_w = F(0);
  const float *cs1_b = F(0), *cs1_w = F(0);
  const float *cs2_b = F(0), *cs2_w = F(0);
  BnRaw dataBnR = { F(0), F(0), F(0), F(0) };
  const float *fcn_b = F(0), *fcn_w = F(0);
  const float *g1_ab[3], *g1_aw[3], *g1_db[3], *g1_dw[3];
  for (int k = 0; k < 3; ++k) g1_ab[k] = F(0);
  for (int k = 0; k < 3; ++k) g1_aw[k] = F(0);
  for (int k = 0; k < 3; ++k) g1_db[k] = F(0);
  for (int k = 0; k < 3; ++k) g1_dw[k] = F(0);
  const float *g2_ab[3], *g2_aw[3], *g2_db[3], *g2_dw[3];
  for (int k = 0; k < 3; ++k) g2_ab[k] = F(0);
  for (int k = 0; k < 3; ++k) g2_aw[k] = F(0);
  for (int k = 0; k < 3; ++k) g2_db[k] = F(0);
  for (int k = 0; k < 3; ++k) g2_dw[k] = F(0);
  struct GcnW { const float* b; BnRaw bn; const float* res_b; BnRaw res_bn;
                const float* res_w; const float* w; };
  GcnW gl[4];
  for (int j = 0; j < 4; ++j) {
    gl[j].b = F(0); gl[j].bn = { F(0), F(0), F(0), F(0) };
    gl[j].res_b = F(0); gl[j].res_bn = { F(0), F(0), F(0), F(0) };
    gl[j].res_w = F(0); gl[j].w = F(0);
  }
  const float* gs1_b = F(0); BnRaw gs1_bnR = { F(0), F(0), F(0), F(0) };
  const float* gs1_w = F(0);
  const float* gs2_b = F(0); BnRaw gs2_bnR = { F(0), F(0), F(0), F(0) };
  const float* gs2_w = F(0);
  struct TcnW { const float* b; BnRaw bn1, bn2; const float* w; };
  TcnW tl[4];
  for (int j = 0; j < 4; ++j) {
    tl[j].b = F(0); tl[j].bn1 = { F(0), F(0), F(0), F(0) };
    tl[j].bn2 = { F(0), F(0), F(0), F(0) }; tl[j].w = F(0);
  }
  TcnW t_mot = { F(0), { F(0),F(0),F(0),F(0) }, { F(0),F(0),F(0),F(0) }, F(0) };
  TcnW t_p1  = { F(0), { F(0),F(0),F(0),F(0) }, { F(0),F(0),F(0),F(0) }, F(0) };
  TcnW t_p2  = { F(0), { F(0),F(0),F(0),F(0) }, { F(0),F(0),F(0),F(0) }, F(0) };
  TcnW t_sh  = { F(0), { F(0),F(0),F(0),F(0) }, { F(0),F(0),F(0),F(0) }, F(0) };

  // ----- workspace carve -----
  float* wsf = (float*)d_ws;
  size_t wpos = 0;
  auto alloc = [&](long long n) { float* p = wsf + wpos; wpos += (size_t)n; return p; };

  float* H0    = alloc(64LL * 3 * 5400);
  float* Abuf  = alloc(64LL * 64 * 5400);
  float* Bbuf  = alloc(64LL * 64 * 5400);
  float* Cbuf  = alloc(64LL * 128 * 5400);
  float* KC    = alloc(8LL * 384 * 5400);
  float* ECH   = alloc(8LL * 128 * 5400);
  float* S3    = alloc(64LL * 3 * 5400);
  float* MOT3  = alloc(64LL * 3 * 5400);
  float* BN3   = alloc(64LL * 3 * 5400);
  float* F0b   = alloc(64LL * 64 * 5400);
  float* F1b   = alloc(64LL * 64 * 2700);
  float* F2b   = alloc(64LL * 64 * 2700);
  float* F3b   = alloc(64LL * 64 * 900);
  float* CAT50 = alloc(64LL * 128 * 900);
  float* RESB  = alloc(64LL * 256 * 900);
  float* HCUR  = alloc(64LL * 256 * 900);
  float* LOGb  = alloc(64 * 324);
  float* SBb   = alloc(64 * 324);
  float* H2    = alloc(64LL * 256 * 18);
  float* Z2    = alloc(64LL * 256 * 18);
  float* Y2    = alloc(64LL * 256 * 18);
  float* POOL  = alloc(32 * 256);

  auto g3 = [](int TV, int C, int NZ) { return dim3((unsigned)((TV + 255) / 256), C, NZ); };
  auto prep = [&](const BnRaw& r, int Cn) {
    float* sc = alloc(Cn); float* sh = alloc(Cn);
    k_bn_prepare<<<(Cn + 63) / 64, 64, 0, stream>>>(r.g, r.b, r.m, r.v, sc, sh, Cn);
    return BnP{ sc, sh };
  };
  auto launch_gemm = [&](const GemmP& p, int gz) {
    dim3 g((p.Cols + 127) / 128, (p.M + 15) / 16, gz);
    if (p.bMode == B_1X1)        k_gemm<B_1X1,   0><<<g, 128, 0, stream>>>(p);
    else if (p.bMode == B_TCONV) k_gemm<B_TCONV, 0><<<g, 128, 0, stream>>>(p);
    else if (p.bMode == B_VLAST && !p.wTrans)
                                 k_gemm<B_VLAST, 0><<<g, 128, 0, stream>>>(p);
    else if (p.bMode == B_VLAST) k_gemm<B_VLAST, 1><<<g, 128, 0, stream>>>(p);
    else                         k_gemm<B_EINSUM,1><<<g, 128, 0, stream>>>(p);
  };

  // ----- folded BN params -----
  BnP dataBn = prep(dataBnR, 54);
  BnP gs1Bn  = prep(gs1_bnR, 64);
  BnP gs2Bn  = prep(gs2_bnR, 128);
  BnP shB1 = prep(t_sh.bn1, 64),  shB2 = prep(t_sh.bn2, 128);
  BnP moB1 = prep(t_mot.bn1, 3),  moB2 = prep(t_mot.bn2, 64);
  BnP p1B1 = prep(t_p1.bn1, 3),   p1B2 = prep(t_p1.bn2, 64);
  BnP p2B1 = prep(t_p2.bn1, 3),   p2B2 = prep(t_p2.bn2, 64);
  BnP tlB1[4], tlB2[4];
  for (int j = 0; j < 4; ++j) { tlB1[j] = prep(tl[j].bn1, 64); tlB2[j] = prep(tl[j].bn2, 64); }
  int glC[4] = { 64, 64, 64, 256 };
  BnP glBn[4], glResBn[4];
  for (int j = 0; j < 4; ++j) {
    glBn[j] = prep(gl[j].bn, glC[j]);
    glResBn[j] = prep(gl[j].res_bn, glC[j]);
  }

  // ----- building blocks -----
  auto gcn_block = [&](const float* xin, int cin, int cmid, const float* w,
                       const float* bias, const float* Aadj, BnP bnp,
                       const float* resPtr, float* outPtr, int Tlen) {
    int TVl = Tlen * 18;
    for (int chk = 0; chk < 8; ++chk) {
      int n0 = chk * 8;
      GemmP pk{}; pk.W = w; pk.bias = bias;
      pk.X = xin + (size_t)n0 * cin * TVl; pk.xStrideN = cin * TVl;
      pk.Out = KC; pk.oStrideN = 3 * cmid * TVl;
      pk.M = 3 * cmid; pk.K = cin; pk.Cols = TVl; pk.bMode = B_1X1; pk.outMode = OUT_ROW;
      launch_gemm(pk, 8);
      GemmP pe{}; pe.W = Aadj; pe.wTrans = 1; pe.M = 18; pe.K = 54;
      pe.Cols = cmid * Tlen; pe.X = KC; pe.xStrideN = 3 * cmid * TVl;
      pe.bMode = B_EINSUM; pe.Tlen = Tlen; pe.cmid = cmid;
      pe.Out = ECH; pe.oStrideN = cmid * TVl; pe.outMode = OUT_COL; pe.oLd = 18;
      launch_gemm(pe, 8);
      size_t off = (size_t)n0 * cmid * TVl;
      k_bn_add_relu<<<g3(TVl, cmid, 8), 256, 0, stream>>>(ECH, resPtr + off, outPtr + off,
                                                          bnp.scale, bnp.shift, TVl);
    }
  };

  auto tcn = [&](const float* xin, int cin, int cout, int Tin, int strideT,
                 const TcnW& tw, BnP b1, BnP b2, float* scratch,
                 float* outPtr, int oStrideN) {
    int Tout = (Tin - 1) / strideT + 1;
    k_bnapply<<<g3(Tin * 18, cin, 64), 256, 0, stream>>>(xin, scratch, b1.scale, b1.shift,
                                                         Tin * 18);
    GemmP p{}; p.W = tw.w; p.bias = tw.b; p.bnscale = b2.scale; p.bnshift = b2.shift;
    p.relu = 1; p.X = scratch; p.xStrideN = cin * Tin * 18;
    p.M = cout; p.K = cin * 3; p.Cols = Tout * 18;
    p.bMode = B_TCONV; p.Tin = Tin; p.strideT = strideT;
    p.Out = outPtr; p.oStrideN = oStrideN; p.outMode = OUT_ROW;
    launch_gemm(p, 64);
  };

  auto attention = [&](const float* h, int Tlen, const float* const* aw,
                       const float* const* ab, const float* const* dw,
                       const float* const* db, float* mbuf, float* zbuf, float* ybuf) {
    int TVl = Tlen * 18, Kred = 128 * Tlen;
    for (int i = 0; i < 3; ++i) {
      GemmP pm{}; pm.W = aw[i]; pm.bias = ab[i]; pm.X = h; pm.xStrideN = 256 * TVl;
      pm.M = 128; pm.K = 256; pm.Cols = TVl; pm.bMode = B_1X1; pm.outMode = OUT_ROW;
      pm.Out = mbuf; pm.oStrideN = 128 * TVl; launch_gemm(pm, 64);
      k_attn_logits<<<64, 352, 0, stream>>>(mbuf, LOGb, Kred, 128 * TVl);
      k_softmax18<<<64, 32, 0, stream>>>(LOGb, SBb);
      GemmP pz{}; pz.W = SBb; pz.wTrans = 1; pz.wStrideN = 324;
      pz.M = 18; pz.K = 18; pz.Cols = 256 * Tlen; pz.X = h; pz.xStrideN = 256 * TVl;
      pz.bMode = B_VLAST; pz.ldB = 18; pz.Out = zbuf; pz.oStrideN = 256 * TVl;
      pz.outMode = OUT_COL; pz.oLd = 18; launch_gemm(pz, 64);
      GemmP py{}; py.W = dw[i]; py.bias = db[i]; py.X = zbuf; py.xStrideN = 256 * TVl;
      py.M = 256; py.K = 256; py.Cols = TVl; py.bMode = B_1X1; py.outMode = OUT_ROW;
      py.Out = ybuf; py.oStrideN = 256 * TVl; py.accum = (i > 0); launch_gemm(py, 64);
    }
  };

  // =================== forward ===================
  k_input_bn<<<g3(5400, 3, 64), 256, 0, stream>>>(IN_x, dataBn.scale, dataBn.shift, H0);

  // conv_shift_1 (3->64)
  { GemmP p{}; p.W = cs1_w; p.bias = cs1_b; p.X = H0; p.xStrideN = 3 * 5400;
    p.M = 64; p.K = 3; p.Cols = 5400; p.bMode = B_1X1; p.outMode = OUT_ROW;
    p.Out = Abuf; p.oStrideN = 64 * 5400; launch_gemm(p, 64); }
  // gcn_shift_1 (res = identity, in-place on Abuf)
  gcn_block(Abuf, 64, 64, gs1_w, gs1_b, A_s1, gs1Bn, Abuf, Abuf, 300);
  // tcn_shift (64->128)
  tcn(Abuf, 64, 128, 300, 1, t_sh, shB1, shB2, Bbuf, Cbuf, 128 * 5400);
  // gcn_shift_2 (in-place on Cbuf)
  gcn_block(Cbuf, 128, 128, gs2_w, gs2_b, A_s2, gs2Bn, Cbuf, Cbuf, 300);
  // conv_shift_2 (128->3)
  { GemmP p{}; p.W = cs2_w; p.bias = cs2_b; p.X = Cbuf; p.xStrideN = 128 * 5400;
    p.M = 3; p.K = 128; p.Cols = 5400; p.bMode = B_1X1; p.outMode = OUT_ROW;
    p.Out = S3; p.oStrideN = 3 * 5400; launch_gemm(p, 64); }
  k_addshift<<<g3(5400, 3, 64), 256, 0, stream>>>(H0, S3);
  k_motion<<<g3(5400, 3, 64), 256, 0, stream>>>(H0, MOT3);

  // pos2 -> Cbuf[:, :64], motion -> Cbuf[:, 64:], pos1 -> Abuf (x_skip)
  tcn(H0,   3, 64, 300, 1, t_p2,  p2B1, p2B2, BN3, Cbuf,              128 * 5400);
  tcn(MOT3, 3, 64, 300, 1, t_mot, moB1, moB2, BN3, Cbuf + 64 * 5400,  128 * 5400);
  tcn(H0,   3, 64, 300, 1, t_p1,  p1B1, p1B2, BN3, Abuf,              64 * 5400);

  // conv_fusion (128->64)
  { GemmP p{}; p.W = fus_w; p.bias = fus_b; p.X = Cbuf; p.xStrideN = 128 * 5400;
    p.M = 64; p.K = 128; p.Cols = 5400; p.bMode = B_1X1; p.outMode = OUT_ROW;
    p.Out = Bbuf; p.oStrideN = 64 * 5400; launch_gemm(p, 64); }

  // tcn_list -> feats (strides 1,2,1,3); Cbuf reused as bn1 scratch
  tcn(Bbuf, 64, 64, 300, 1, tl[0], tlB1[0], tlB2[0], Cbuf, F0b, 64 * 5400);
  tcn(F0b,  64, 64, 300, 2, tl[1], tlB1[1], tlB2[1], Cbuf, F1b, 64 * 2700);
  tcn(F1b,  64, 64, 150, 1, tl[2], tlB1[2], tlB2[2], Cbuf, F2b, 64 * 2700);
  tcn(F2b,  64, 64, 150, 3, tl[3], tlB1[3], tlB2[3], Cbuf, F3b, 64 * 900);

  // gcn loop @ T=50; feats reversed = [f3, f2, f1, f0]
  const float* fsrc[4] = { F3b, F2b, F1b, F0b };
  int          frat[4] = { 1, 3, 3, 6 };
  int          fist[4] = { 64 * 900, 64 * 2700, 64 * 2700, 64 * 5400 };
  for (int j = 0; j < 4; ++j) {
    const float* hsrc = (j == 0) ? Abuf : HCUR;
    int hrat = (j == 0) ? 6 : 1;
    int histr = (j == 0) ? 64 * 5400 : 64 * 900;
    k_scaleT<<<g3(900, 64, 64), 256, 0, stream>>>(hsrc, CAT50, 50, hrat, histr, 128 * 900);
    k_scaleT<<<g3(900, 64, 64), 256, 0, stream>>>(fsrc[j], CAT50 + 64 * 900, 50, frat[j],
                                                  fist[j], 128 * 900);
    { GemmP p{}; p.W = gl[j].res_w; p.bias = gl[j].res_b;
      p.bnscale = glResBn[j].scale; p.bnshift = glResBn[j].shift;
      p.X = CAT50; p.xStrideN = 128 * 900; p.M = glC[j]; p.K = 128; p.Cols = 900;
      p.bMode = B_1X1; p.outMode = OUT_ROW; p.Out = RESB;
      p.oStrideN = glC[j] * 900; launch_gemm(p, 64); }
    gcn_block(CAT50, 128, glC[j], gl[j].w, gl[j].b, As[j], glBn[j], RESB, HCUR, 50);
  }

  // gather1 (c=256, T=50): mbuf=CAT50, z=Cbuf, y=Bbuf
  attention(HCUR, 50, g1_aw, g1_ab, g1_dw, g1_db, CAT50, Cbuf, Bbuf);
  k_maxT<<<dim3(1, 256, 64), 32, 0, stream>>>(Bbuf, H2, 50);
  // gather2 (T=1)
  attention(H2, 1, g2_aw, g2_ab, g2_dw, g2_db, CAT50, Z2, Y2);
  k_pool_mv<<<(32 * 256 + 255) / 256, 256, 0, stream>>>(Y2, POOL);

  // final FC: out[32,400] = POOL[32,256] @ W^T + b
  { GemmP p{}; p.W = fcn_w; p.bias = fcn_b; p.X = POOL;
    p.M = 400; p.K = 256; p.Cols = 32; p.bMode = B_VLAST; p.ldB = 256;
    p.outMode = OUT_COL; p.oLd = 400; p.Out = (float*)d_out; launch_gemm(p, 1); }
}
